// PanelSegRetinaNet_10746008175165
// MI455X (gfx1250) — compile-verified, matrix operand order checked by reading, not run on hardware
//
#include <hip/hip_runtime.h>

typedef _Float16 half8 __attribute__((ext_vector_type(8)));
typedef _Float16 v16h  __attribute__((ext_vector_type(16)));
typedef float    v8f   __attribute__((ext_vector_type(8)));
typedef int      v4i   __attribute__((vector_size(16)));   // matches async-LDS builtin param

#define NEG_INF (-3.402823466e38f)

#define GLOBAL_AS __attribute__((address_space(1)))
#define LDS_AS    __attribute__((address_space(3)))

#if defined(__has_builtin)
#if __has_builtin(__builtin_amdgcn_global_load_async_to_lds_b128)
#define HAVE_ASYNC_LDS 1
#endif
#endif
#ifndef HAVE_ASYNC_LDS
#define HAVE_ASYNC_LDS 0
#endif

union FragH { v16h v; half8 h[2]; };

// ---------------------------------------------------------------------------
// fp32 -> f16 conversion (activations)
// ---------------------------------------------------------------------------
__global__ void cvt_f32_f16(const float* __restrict__ in, _Float16* __restrict__ out, int n) {
  int i = blockIdx.x * blockDim.x + threadIdx.x;
  if (i < n) out[i] = (_Float16)in[i];
}

// fp32 -> f16 weight conversion, OC zero-padded, K reordered so that
//   k = (ky*3+kx)*256 + ic    (input is OIHW: k_in = ic*9 + ky*3 + kx)
// A 32-wide K slab then has uniform (ky,kx) and channel-contiguous ic.
__global__ void cvt_w_perm(const float* __restrict__ in, _Float16* __restrict__ out,
                           int oc_real, int oc_pad) {
  int i = blockIdx.x * blockDim.x + threadIdx.x;
  int total = oc_pad * 2304;
  if (i >= total) return;
  int oc = i / 2304;
  int t  = i - oc * 2304;
  int r  = t >> 8;      // kernel position 0..8
  int ic = t & 255;
  float v = (oc < oc_real) ? in[oc * 2304 + ic * 9 + r] : 0.f;
  out[i] = (_Float16)v;
}

// ---------------------------------------------------------------------------
// Implicit-GEMM 3x3 conv (IC=256, SAME padding), f16 in, f32 accum via WMMA.
//   M = OC (padded to WAVES_M*MT*16), N = H*W, K = 9*256 = 2304.
//   Block = 256 threads = 8 wave32s arranged WAVES_M x WAVES_N.
//   Double-buffered LDS tiles; weight tile staged with async global->LDS
//   (ASYNCcnt) when available so global latency overlaps WMMA.
// ---------------------------------------------------------------------------
template<int WAVES_M, int WAVES_N, int MT, bool RELU, bool OUT16>
__global__ __launch_bounds__(256) void conv3x3_wmma(
    const _Float16* __restrict__ X,     // [256][H][W] f16
    const _Float16* __restrict__ Wt,    // [OCPAD][2304] f16, k=(ky*3+kx)*256+ic
    const float*    __restrict__ bias,  // [OCreal]
    void*           __restrict__ outp,  // [OCreal][H*W], f16 or f32
    int Hd, int Wd, int OCreal)
{
  constexpr int OCPAD = WAVES_M * MT * 16;
  constexpr int NTILE = WAVES_N * 16;
  constexpr int KSTEP = 32;
  constexpr int KTOT  = 9 * 256;
  constexpr int NIT   = KTOT / KSTEP;           // 72
  constexpr int ACH   = OCPAD * 4;              // 16B chunks per A tile
  // per-thread async ops per tile (only usable as a wait target if uniform)
  constexpr int AWAIT = (ACH % 256 == 0) ? (ACH / 256) : 0;

  __shared__ _Float16 Ash[2][OCPAD * KSTEP];    // weight tiles  [OCPAD][32]
  __shared__ _Float16 Bsh[2][NTILE * KSTEP];    // im2col tiles  [NTILE][32]

  const int tid  = threadIdx.x;
  const int wave = tid >> 5;
  const int lane = tid & 31;
  const int wm   = wave % WAVES_M;
  const int wn   = wave / WAVES_M;
  const int ml   = lane & 15;
  const int kh   = (lane >> 4) * 8;             // K-half per WMMA 16-bit layout

  const int HW = Hd * Wd;
  const int n0 = blockIdx.x * NTILE;

  // ---- per-thread spatial position for the im2col gather (hoisted) ----
  const int nl  = tid % NTILE;                  // constant across this thread's slab work
  const int n   = n0 + nl;
  const int yy  = (n < HW) ? (n / Wd) : 0;
  const int xx  = (n < HW) ? (n - yy * Wd) : 0;

  auto stageA = [&](int buf, int k0) {
    for (int i = tid; i < ACH; i += 256) {
      int oc = i >> 2;
      int kc = (i & 3) * 8;
#if HAVE_ASYNC_LDS
      __builtin_amdgcn_global_load_async_to_lds_b128(
          (GLOBAL_AS v4i*)(Wt + oc * KTOT + k0 + kc),
          (LDS_AS v4i*)&Ash[buf][oc * KSTEP + kc], 0, 0);
#else
      *(half8*)&Ash[buf][oc * KSTEP + kc] = *(const half8*)&Wt[oc * KTOT + k0 + kc];
#endif
    }
#if !HAVE_ASYNC_LDS
    if (k0 + KSTEP < KTOT && tid < OCPAD)
      __builtin_prefetch(&Wt[tid * KTOT + k0 + KSTEP], 0, 1);
#endif
  };

  auto stageB = [&](int buf, int k0) {
    const int r   = k0 >> 8;                    // uniform kernel position
    const int ky  = r / 3;
    const int kx  = r - ky * 3;
    const int ic0 = k0 & 255;
    const int y   = yy + ky - 1;
    const int x   = xx + kx - 1;
    const bool inb = (n < HW) && (y >= 0) && (y < Hd) && (x >= 0) && (x < Wd);
    const _Float16* src = X + (size_t)y * Wd + x;   // only dereferenced if inb
    for (int kk = tid / NTILE; kk < KSTEP; kk += (256 / NTILE)) {
      _Float16 v = inb ? src[(size_t)(ic0 + kk) * HW] : (_Float16)0.f;
      Bsh[buf][nl * KSTEP + kk] = v;            // adjacent threads: adjacent x -> coalesced loads
    }
  };

  v8f acc[MT] = {};

  stageA(0, 0);
  stageB(0, 0);

  for (int it = 0; it < NIT; ++it) {
    const int cur = it & 1;
    const int nxt = cur ^ 1;
    if (it + 1 < NIT) {                          // prefetch next tile into other buffer
      stageA(nxt, (it + 1) * KSTEP);
      stageB(nxt, (it + 1) * KSTEP);
    }
#if HAVE_ASYNC_LDS
    // async loads complete in order: allowing AWAIT outstanding (the just-issued
    // next tile) guarantees the current tile has landed in LDS.
    if (it + 1 < NIT) {
      asm volatile("s_wait_asynccnt %0" :: "i"(AWAIT) : "memory");
    } else {
      asm volatile("s_wait_asynccnt 0" ::: "memory");
    }
#endif
    __syncthreads();

    FragH bfrag;
    {
      const int bl = wn * 16 + ml;
      bfrag.h[0] = *(const half8*)&Bsh[cur][bl * KSTEP + kh];
      bfrag.h[1] = *(const half8*)&Bsh[cur][bl * KSTEP + kh + 16];
    }
#pragma unroll
    for (int t = 0; t < MT; ++t) {
      FragH afrag;
      const int row = (wm * MT + t) * 16 + ml;
      afrag.h[0] = *(const half8*)&Ash[cur][row * KSTEP + kh];
      afrag.h[1] = *(const half8*)&Ash[cur][row * KSTEP + kh + 16];
      acc[t] = __builtin_amdgcn_wmma_f32_16x16x32_f16(
          false, afrag.v, false, bfrag.v, (short)0, acc[t], false, false);
    }
    __syncthreads();
  }

  // ---- epilogue: bias (+ReLU), store f16 or f32 ----
  const int ncol = n0 + wn * 16 + ml;
#pragma unroll
  for (int t = 0; t < MT; ++t) {
#pragma unroll
    for (int r = 0; r < 8; ++r) {
      int m = (wm * MT + t) * 16 + r + ((lane >> 4) << 3);
      if (m < OCreal && ncol < HW) {
        float val = acc[t][r] + bias[m];
        if (RELU) val = fmaxf(val, 0.f);
        if constexpr (OUT16) ((_Float16*)outp)[m * HW + ncol] = (_Float16)val;
        else                 ((float*)outp)[m * HW + ncol] = val;
      }
    }
  }
}

// ---------------------------------------------------------------------------
// sigmoid + score-threshold mask + top-k selection (one workgroup per level).
// Tie-break toward lower index to mirror lax.top_k ordering.
// ---------------------------------------------------------------------------
__global__ __launch_bounds__(256) void select_topk(
    const float* __restrict__ logits,   // [9][HW]
    float* __restrict__ scm,            // [HW*9] scratch
    int HW, int k,
    float* __restrict__ cand_s, int* __restrict__ cand_a, int off)
{
  const int tid = threadIdx.x;
  const int HWA = HW * 9;
  for (int i = tid; i < HWA; i += 256) {
    int a = i % 9;
    int n = i / 9;
    float s = 1.f / (1.f + __expf(-logits[a * HW + n]));
    scm[i] = (s > 0.05f) ? s : -1.f;
  }
  __syncthreads();
  __shared__ float sv[256];
  __shared__ int   si[256];
  for (int it = 0; it < k; ++it) {
    float bv = NEG_INF; int bi = 0x7fffffff;
    for (int i = tid; i < HWA; i += 256) {
      float v = scm[i];
      if (v > bv || (v == bv && i < bi)) { bv = v; bi = i; }
    }
    sv[tid] = bv; si[tid] = bi;
    __syncthreads();
    for (int s = 128; s > 0; s >>= 1) {
      if (tid < s) {
        if (sv[tid+s] > sv[tid] || (sv[tid+s] == sv[tid] && si[tid+s] < si[tid])) {
          sv[tid] = sv[tid+s]; si[tid] = si[tid+s];
        }
      }
      __syncthreads();
    }
    if (tid == 0) {
      cand_s[off + it] = sv[0];
      cand_a[off + it] = si[0];
      scm[si[0]] = -3.f;   // below the -1 mask so -1 entries pad in index order
    }
    __syncthreads();
  }
}

// ---------------------------------------------------------------------------
// Box decode (anchors generated analytically; NUM_CLASSES == 1 -> class 0)
// ---------------------------------------------------------------------------
__global__ void decode_boxes(const float* __restrict__ deltas,  // [36][HW]
                             const int* __restrict__ cand_a,
                             float* __restrict__ boxes, int off, int k,
                             int HW, int Wd, float stride, float base)
{
  int j = blockIdx.x * blockDim.x + threadIdx.x;
  if (j >= k) return;
  int hwa = cand_a[off + j];
  int a = hwa % 9;
  int n = hwa / 9;
  int y = n / Wd;
  int x = n - y * Wd;
  int sidx = a / 3;
  int ridx = a - sidx * 3;
  float scale = exp2f((float)sidx * (1.f / 3.f));
  float ratio = (ridx == 0) ? 0.5f : ((ridx == 1) ? 1.f : 2.f);
  float bs = base * scale;
  float wA = sqrtf(bs * bs / ratio);
  float hA = wA * ratio;
  float cxa = (float)x * stride;
  float cya = (float)y * stride;
  float dx = deltas[(a * 4 + 0) * HW + n];
  float dy = deltas[(a * 4 + 1) * HW + n];
  float dw = deltas[(a * 4 + 2) * HW + n];
  float dh = deltas[(a * 4 + 3) * HW + n];
  const float CLAMP = 4.135166556742356f;   // log(1000/16)
  dw = fminf(dw, CLAMP); dh = fminf(dh, CLAMP);
  float cx = dx * wA + cxa;
  float cy = dy * hA + cya;
  float w = __expf(dw) * wA;
  float h = __expf(dh) * hA;
  float* bo = boxes + (size_t)(off + j) * 4;
  bo[0] = cx - 0.5f * w; bo[1] = cy - 0.5f * h;
  bo[2] = cx + 0.5f * w; bo[3] = cy + 0.5f * h;
}

__device__ __forceinline__ float iou4(const float4& a, const float4& b) {
  float areaA = (a.z - a.x) * (a.w - a.y);
  float areaB = (b.z - b.x) * (b.w - b.y);
  float lx = fmaxf(a.x, b.x), ly = fmaxf(a.y, b.y);
  float rx = fminf(a.z, b.z), ry = fminf(a.w, b.w);
  float iw = fmaxf(rx - lx, 0.f), ih = fmaxf(ry - ly, 0.f);
  float inter = iw * ih;
  return inter / (areaA + areaB - inter + 1e-9f);
}

// ---------------------------------------------------------------------------
// Greedy NMS (single workgroup): process in descending score order, suppress
// vs previously kept, emit [100][6] = box, score, class (pad score = -1).
// ---------------------------------------------------------------------------
__global__ __launch_bounds__(256) void nms_kernel(
    const float* __restrict__ scores, const float* __restrict__ boxes,
    float* __restrict__ rem, int* __restrict__ order, int* __restrict__ keepf,
    int* __restrict__ klist, float* __restrict__ out, int TOT)
{
  const int tid = threadIdx.x;
  __shared__ float sv[256];
  __shared__ int   si[256];
  __shared__ int   s_kc, s_np, s_sup;
  for (int i = tid; i < TOT; i += 256) rem[i] = scores[i];
  if (tid == 0) { s_kc = 0; s_np = 0; }
  __syncthreads();
  const float4* b4 = (const float4*)boxes;

  for (int it = 0; it < TOT; ++it) {
    float bv = NEG_INF; int bi = 0x7fffffff;
    for (int i = tid; i < TOT; i += 256) {
      float v = rem[i];
      if (v > bv || (v == bv && i < bi)) { bv = v; bi = i; }
    }
    sv[tid] = bv; si[tid] = bi;
    __syncthreads();
    for (int s = 128; s > 0; s >>= 1) {
      if (tid < s) {
        if (sv[tid+s] > sv[tid] || (sv[tid+s] == sv[tid] && si[tid+s] < si[tid])) {
          sv[tid] = sv[tid+s]; si[tid] = si[tid+s];
        }
      }
      __syncthreads();
    }
    float v = sv[0]; int idx = si[0];
    if (v <= 0.f) break;               // remaining are invalid (-1 padding)
    if (tid == 0) { rem[idx] = NEG_INF; s_sup = 0; }
    __syncthreads();
    float4 bb = b4[idx];
    int kc = s_kc;
    int sup = 0;
    for (int j = tid; j < kc; j += 256)
      if (iou4(bb, b4[klist[j]]) > 0.5f) sup = 1;
    if (sup) atomicOr(&s_sup, 1);
    __syncthreads();
    if (tid == 0) {
      int np = s_np;
      order[np] = idx;
      int kp = (s_sup == 0) ? 1 : 0;
      keepf[np] = kp;
      if (kp) { klist[s_kc] = idx; s_kc = s_kc + 1; }
      s_np = np + 1;
    }
    __syncthreads();
  }

  __syncthreads();
  if (tid == 0) {
    // remaining (all score <= 0) pad in ascending index order (top_k tie order)
    int np = s_np;
    for (int i = 0; i < TOT && np < TOT; ++i)
      if (rem[i] != NEG_INF) { order[np] = i; keepf[np] = 0; np++; }
    s_np = np;
    // kept entries first (score desc), then non-kept (score = -1)
    int p = 0;
    for (int pass = 0; pass < 2 && p < 100; ++pass) {
      for (int i = 0; i < s_np && p < 100; ++i) {
        if ((keepf[i] != 0) == (pass == 0)) {
          int id = order[i];
          float4 bb = b4[id];
          out[p*6+0] = bb.x; out[p*6+1] = bb.y;
          out[p*6+2] = bb.z; out[p*6+3] = bb.w;
          out[p*6+4] = (pass == 0) ? scores[id] : -1.f;
          out[p*6+5] = 0.f;
          ++p;
        }
      }
    }
    while (p < 100) {
      out[p*6+0] = 0.f; out[p*6+1] = 0.f; out[p*6+2] = 0.f; out[p*6+3] = 0.f;
      out[p*6+4] = -1.f; out[p*6+5] = 0.f; ++p;
    }
  }
}

// ---------------------------------------------------------------------------
// Host orchestration
// ---------------------------------------------------------------------------
extern "C" void kernel_launch(void* const* d_in, const int* in_sizes, int n_in,
                              void* d_out, int out_size, void* d_ws, size_t ws_size,
                              hipStream_t stream) {
  (void)in_sizes; (void)n_in; (void)out_size; (void)ws_size;

  const float* feats[5] = {(const float*)d_in[0], (const float*)d_in[1],
                           (const float*)d_in[2], (const float*)d_in[3],
                           (const float*)d_in[4]};
  const float* cls_w   = (const float*)d_in[5];
  const float* cls_b   = (const float*)d_in[6];
  const float* bbox_w  = (const float*)d_in[7];
  const float* bbox_b  = (const float*)d_in[8];
  const float* score_w = (const float*)d_in[9];
  const float* score_b = (const float*)d_in[10];
  const float* pred_w  = (const float*)d_in[11];
  const float* pred_b  = (const float*)d_in[12];

  static const int   Hs[5]  = {100, 50, 25, 13, 7};
  static const int   Wds[5] = {100, 50, 25, 13, 7};
  static const float Sts[5] = {8.f, 16.f, 32.f, 64.f, 128.f};
  static const float Bas[5] = {32.f, 64.f, 128.f, 256.f, 512.f};

  char* wsb = (char*)d_ws;
  size_t off = 0;
  auto carve = [&](size_t bytes) -> void* {
    void* p = wsb + off;
    off += (bytes + 255) & ~(size_t)255;
    return p;
  };

  const size_t LW = 256ull * 2304;          // halves per 256-OC layer
  _Float16* clsw16   = (_Float16*)carve(4 * LW * 2);
  _Float16* bboxw16  = (_Float16*)carve(4 * LW * 2);
  _Float16* scorew16 = (_Float16*)carve(16ull * 2304 * 2);
  _Float16* predw16  = (_Float16*)carve(48ull * 2304 * 2);
  _Float16* feat16   = (_Float16*)carve(256ull * 10000 * 2);
  _Float16* t0       = (_Float16*)carve(256ull * 10000 * 2);
  _Float16* t1       = (_Float16*)carve(256ull * 10000 * 2);
  float* score32 = (float*)carve(9ull  * 10000 * 4);
  float* pred32  = (float*)carve(36ull * 10000 * 4);
  float* scm     = (float*)carve(90000ull * 4);
  float* cand_s  = (float*)carve(4441ull * 4);
  int*   cand_a  = (int*)  carve(4441ull * 4);
  float* cand_bx = (float*)carve(4441ull * 16);
  float* rem     = (float*)carve(4441ull * 4);
  int*   order   = (int*)  carve(4441ull * 4);
  int*   keepf   = (int*)  carve(4441ull * 4);
  int*   klist   = (int*)  carve(4441ull * 4);

  // ---- weight prep: fp32 -> f16, K-permute, heads zero-padded ----
  {
    int n = (int)(4 * LW);
    cvt_w_perm<<<(n + 255) / 256, 256, 0, stream>>>(cls_w,  clsw16,  1024, 1024);
    cvt_w_perm<<<(n + 255) / 256, 256, 0, stream>>>(bbox_w, bboxw16, 1024, 1024);
    cvt_w_perm<<<(16 * 2304 + 255) / 256, 256, 0, stream>>>(score_w, scorew16, 9, 16);
    cvt_w_perm<<<(48 * 2304 + 255) / 256, 256, 0, stream>>>(pred_w,  predw16, 36, 48);
  }

  int coff = 0;
  for (int l = 0; l < 5; ++l) {
    const int H = Hs[l], W = Wds[l], HW = H * W;
    cvt_f32_f16<<<(256 * HW + 255) / 256, 256, 0, stream>>>(feats[l], feat16, 256 * HW);

    const int gN = (HW + 31) / 32;      // subnet: N-tile 32
    const int gH = (HW + 127) / 128;    // heads:  N-tile 128

    // ---- cls tower: feat -> t0 -> t1 -> t0 -> t1, then score head ----
    conv3x3_wmma<4,2,4,true,true><<<gN,256,0,stream>>>(feat16, clsw16 + 0*LW, cls_b + 0*256, (void*)t0, H, W, 256);
    conv3x3_wmma<4,2,4,true,true><<<gN,256,0,stream>>>(t0,     clsw16 + 1*LW, cls_b + 1*256, (void*)t1, H, W, 256);
    conv3x3_wmma<4,2,4,true,true><<<gN,256,0,stream>>>(t1,     clsw16 + 2*LW, cls_b + 2*256, (void*)t0, H, W, 256);
    conv3x3_wmma<4,2,4,true,true><<<gN,256,0,stream>>>(t0,     clsw16 + 3*LW, cls_b + 3*256, (void*)t1, H, W, 256);
    conv3x3_wmma<1,8,1,false,false><<<gH,256,0,stream>>>(t1, scorew16, score_b, (void*)score32, H, W, 9);

    // ---- bbox tower, then pred head ----
    conv3x3_wmma<4,2,4,true,true><<<gN,256,0,stream>>>(feat16, bboxw16 + 0*LW, bbox_b + 0*256, (void*)t0, H, W, 256);
    conv3x3_wmma<4,2,4,true,true><<<gN,256,0,stream>>>(t0,     bboxw16 + 1*LW, bbox_b + 1*256, (void*)t1, H, W, 256);
    conv3x3_wmma<4,2,4,true,true><<<gN,256,0,stream>>>(t1,     bboxw16 + 2*LW, bbox_b + 2*256, (void*)t0, H, W, 256);
    conv3x3_wmma<4,2,4,true,true><<<gN,256,0,stream>>>(t0,     bboxw16 + 3*LW, bbox_b + 3*256, (void*)t1, H, W, 256);
    conv3x3_wmma<1,8,3,false,false><<<gH,256,0,stream>>>(t1, predw16, pred_b, (void*)pred32, H, W, 36);

    // ---- per-level top-k + decode ----
    const int HWA = HW * 9;
    const int k = (HWA < 1000) ? HWA : 1000;
    select_topk<<<1, 256, 0, stream>>>(score32, scm, HW, k, cand_s, cand_a, coff);
    decode_boxes<<<(k + 255) / 256, 256, 0, stream>>>(pred32, cand_a, cand_bx, coff, k,
                                                      HW, W, Sts[l], Bas[l]);
    coff += k;
  }

  nms_kernel<<<1, 256, 0, stream>>>(cand_s, cand_bx, rem, order, keepf, klist,
                                    (float*)d_out, coff);
}
